// LatentODE_52948356825366
// MI455X (gfx1250) — matrix operand order repaired and checked by reading, not compiled, not run on hardware
//
#include <hip/hip_runtime.h>
#include <hip/hip_bf16.h>
#include <math.h>

// ---------------------------------------------------------------------------
// LatentODE on gfx1250 (MI455X), wave32 + WMMA.
//  - enc_kernel: backward GRU scan + z0/h0/y0  (WMMA f32_16x16x32_f16)
//  - ode_kernel: 64 save steps x 2 substeps x 6 Tsit5 stages; each stage's
//    MLP is 3 WMMA GEMMs over a 16-batch tile, all weights resident in LDS.
// Round 2: scalarized wave-tile loops (readfirstlane) so WMMA loops lower to
// scalar control flow (EXEC stays all-ones, no per-lane mask churn), and fast
// hardware transcendentals (v_rcp/v_exp/v_log) in the epilogues.
// ---------------------------------------------------------------------------

typedef _Float16 v16h __attribute__((ext_vector_type(16)));
typedef _Float16 v8h  __attribute__((ext_vector_type(8)));
typedef float    v8f  __attribute__((ext_vector_type(8)));

#define NB    16     // batch tile per workgroup (WMMA N)
#define NTHR  256    // 8 wave32s
#define NWAVE 8
#define TT    64
#define HH    128
#define WW    256
#define DD    32

__constant__ float c_C[6] = {0.f, 0.161f, 0.327f, 0.9f, 0.9800255409045097f, 1.f};
__constant__ float c_A[6][5] = {
  {0.f, 0.f, 0.f, 0.f, 0.f},
  {0.161f, 0.f, 0.f, 0.f, 0.f},
  {-0.008480655492356989f, 0.335480655492357f, 0.f, 0.f, 0.f},
  {2.8971530571054935f, -6.359448489975075f, 4.3622954328695815f, 0.f, 0.f},
  {5.325864828439257f, -11.748883564062828f, 7.4955393428898365f, -0.09249506636175525f, 0.f},
  {5.86145544294642f, -12.92096931784711f, 8.159367898576159f, -0.071584973281401f, -0.028269050394068383f}
};
__constant__ float c_B[6] = {0.09646076681806523f, 0.01f, 0.4798896504144996f,
                             1.379008574103742f, -3.290069515436081f, 2.324710524099774f};

// ---- fast transcendentals (v_exp_f32 / v_log_f32 / v_rcp_f32) ----
__device__ __forceinline__ float fast_rcp(float x)  { return __builtin_amdgcn_rcpf(x); }
__device__ __forceinline__ float sigm(float x) {
  return fast_rcp(1.0f + __expf(-x));
}
__device__ __forceinline__ float softplus(float x) {
  return fmaxf(x, 0.0f) + __logf(1.0f + __expf(-fabsf(x)));
}
__device__ __forceinline__ float fast_tanh(float x) {
  float xc = fminf(fmaxf(x, -15.0f), 15.0f);
  float e  = __expf(2.0f * xc);
  return (e - 1.0f) * fast_rcp(e + 1.0f);
}

// A fragment: 16x32 f16 tile of W[M][Kpad] (row-major). ISA A-layout:
// lanes 0-15 hold M=lane, K = kBase+{sel*8..sel*8+7} and +16; sel = lane>>4.
__device__ __forceinline__ v16h frag_a(const _Float16* W, int Kpad, int mBase, int kBase) {
  int lane = threadIdx.x & 31;
  const _Float16* p = W + (mBase + (lane & 15)) * Kpad + kBase + ((lane >> 4) << 3);
  v8h lo = *(const v8h*)p;
  v8h hi = *(const v8h*)(p + 16);
  return __builtin_shufflevector(lo, hi, 0,1,2,3,4,5,6,7,8,9,10,11,12,13,14,15);
}
// B fragment: 32x16 tile; activations stored column-major xT[N][Kpad] so each
// lane reads K = kBase+{sel*8..sel*8+7} and +16 (two ds_load_b128), matching A.
__device__ __forceinline__ v16h frag_b(const _Float16* xT, int Kpad, int kBase) {
  int lane = threadIdx.x & 31;
  const _Float16* p = xT + (lane & 15) * Kpad + kBase + ((lane >> 4) << 3);
  v8h lo = *(const v8h*)p;
  v8h hi = *(const v8h*)(p + 16);
  return __builtin_shufflevector(lo, hi, 0,1,2,3,4,5,6,7,8,9,10,11,12,13,14,15);
}
__device__ __forceinline__ v8f wmma_tile(const _Float16* W, int Kpad, int mBase,
                                         const _Float16* xT, int Ktiles) {
  v8f acc = {};
  #pragma unroll
  for (int kt = 0; kt < Ktiles; ++kt) {
    v16h a = frag_a(W, Kpad, mBase, kt * 32);
    v16h b = frag_b(xT, Kpad, kt * 32);
    acc = __builtin_amdgcn_wmma_f32_16x16x32_f16(false, a, false, b, (short)0, acc,
                                                 false, false);
  }
  return acc;
}

#define CARVE(T, name, n) T* name = (T*)(smem + _off); _off += (((size_t)(n) * sizeof(T)) + 31) & ~(size_t)31;

// ======================= GRU encoder =======================
#define ENC_SMEM 230496u

__global__ __launch_bounds__(NTHR)
void enc_kernel(const float* __restrict__ ts, const float* __restrict__ ys,
                const float* __restrict__ wih, const float* __restrict__ whh,
                const float* __restrict__ gb,  const float* __restrict__ gbn,
                const float* __restrict__ htlw, const float* __restrict__ htlb,
                const float* __restrict__ ltiw, const float* __restrict__ ltib,
                const float* __restrict__ lthw, const float* __restrict__ lthb,
                float* __restrict__ out_z, float* __restrict__ st0)
{
  extern __shared__ char smem[];
  size_t _off = 0;
  CARVE(_Float16, sWih,  384*32);   // gru_wih cols 1..32 (t folded out)
  CARVE(float,    sWih0, 384);      // gru_wih column 0
  CARVE(_Float16, sWhh,  384*128);
  CARVE(float,    sGb,   384);
  CARVE(float,    sGbn,  128);
  CARVE(float,    sHid,  NB*128);
  CARVE(_Float16, sHidH, NB*128);   // fp16 copy, column-major [n][128]
  CARVE(_Float16, sXT,   NB*32);    // input y tile, column-major [n][32]
  CARVE(float,    sGi,   NB*384);   // Wih@x results, [n][384]
  CARVE(float,    sGh,   NB*384);   // Whh@h results, [n][384]
  CARVE(_Float16, sHtlw, 64*128);
  CARVE(float,    sHtlb, 64);
  CARVE(_Float16, sLthw, 128*64);
  CARVE(float,    sLthb, 128);
  CARVE(float,    sLtiw, 5*64);
  CARVE(float,    sLtib, 5);
  CARVE(float,    sZ0,   NB*64);
  CARVE(_Float16, sZ0H,  NB*64);
  CARVE(float,    sTs,   NB);
  CARVE(float,    sLog,  NB*8);

  const int tid  = threadIdx.x, lane = tid & 31;
  const int wave = __builtin_amdgcn_readfirstlane(tid >> 5);  // scalar wave id
  const int b0 = blockIdx.x * NB;

  // ---- stage weights into LDS (f32 -> f16) ----
  for (int i = tid; i < 384; i += NTHR) { sWih0[i] = wih[i*33]; sGb[i] = gb[i]; }
  for (int i = tid; i < 384*32; i += NTHR) {
    int m = i >> 5, k = i & 31;
    sWih[i] = (_Float16)wih[m*33 + 1 + k];
  }
  for (int i = tid; i < 384*128; i += NTHR) sWhh[i] = (_Float16)whh[i];
  for (int i = tid; i < 128; i += NTHR) { sGbn[i] = gbn[i]; sLthb[i] = lthb[i]; }
  for (int i = tid; i < 64*128; i += NTHR) sHtlw[i] = (_Float16)htlw[i];
  for (int i = tid; i < 128*64; i += NTHR) sLthw[i] = (_Float16)lthw[i];
  for (int i = tid; i < 64; i += NTHR) sHtlb[i] = htlb[i];
  for (int i = tid; i < 5*64; i += NTHR) sLtiw[i] = ltiw[i];
  if (tid < 5) sLtib[tid] = ltib[tid];
  for (int i = tid; i < NB*128; i += NTHR) { sHid[i] = 0.0f; sHidH[i] = (_Float16)0.0f; }
  __syncthreads();

  // ---- backward scan over time ----
  for (int s = TT - 1; s >= 0; --s) {
    for (int i = tid; i < NB*32; i += NTHR) {
      int n = i >> 5, k = i & 31;
      sXT[i] = (_Float16)ys[((size_t)(b0 + n) * TT + s) * DD + k];
    }
    if (tid < NB) sTs[tid] = ts[(size_t)(b0 + tid) * TT + s];
    __syncthreads();

    for (int mt = wave; mt < 24; mt += NWAVE) {            // 384 rows = 24 tiles
      v8f ai = wmma_tile(sWih, 32,  mt * 16, sXT,   1);    // K=32
      v8f ah = wmma_tile(sWhh, 128, mt * 16, sHidH, 4);    // K=128
      int n = lane & 15, row0 = mt * 16 + ((lane >> 4) << 3);
      *(v8f*)(sGi + n * 384 + row0) = ai;
      *(v8f*)(sGh + n * 384 + row0) = ah;
    }
    __syncthreads();

    for (int i = tid; i < NB*128; i += NTHR) {
      int n = i >> 7, j = i & 127;
      float tn = sTs[n];
      float ir = sGi[n*384 + j]       + sGb[j]       + tn * sWih0[j];
      float iz = sGi[n*384 + 128 + j] + sGb[128 + j] + tn * sWih0[128 + j];
      float in_= sGi[n*384 + 256 + j] + sGb[256 + j] + tn * sWih0[256 + j];
      float hr = sGh[n*384 + j];
      float hz = sGh[n*384 + 128 + j];
      float hn = sGh[n*384 + 256 + j];
      float r  = sigm(ir + hr);
      float z  = sigm(iz + hz);
      float nn = fast_tanh(in_ + r * (hn + sGbn[j]));
      float h  = nn + z * (sHid[i] - nn);
      sHid[i]  = h;
      sHidH[i] = (_Float16)h;
    }
    __syncthreads();
  }

  // ---- z0 = htl_w @ hid + htl_b ----
  for (int mt = wave; mt < 4; mt += NWAVE) {
    v8f a = wmma_tile(sHtlw, 128, mt * 16, sHidH, 4);
    int n = lane & 15, row0 = mt * 16 + ((lane >> 4) << 3);
    #pragma unroll
    for (int v = 0; v < 8; ++v) {
      float zv = a[v] + sHtlb[row0 + v];
      sZ0[n*64 + row0 + v]  = zv;
      sZ0H[n*64 + row0 + v] = (_Float16)zv;
      out_z[(size_t)(b0 + n) * 64 + row0 + v] = zv;
    }
  }
  __syncthreads();

  // ---- h0 = lth_w @ z0 + lth_b  -> state0[5:133] ----
  for (int mt = wave; mt < 8; mt += NWAVE) {
    v8f a = wmma_tile(sLthw, 64, mt * 16, sZ0H, 2);
    int n = lane & 15, row0 = mt * 16 + ((lane >> 4) << 3);
    #pragma unroll
    for (int v = 0; v < 8; ++v)
      st0[(size_t)(b0 + n) * 136 + 5 + row0 + v] = a[v] + sLthb[row0 + v];
  }
  // ---- y0 = softmax(lti_w @ z0 + lti_b) -> state0[0:5] ----
  if (tid < 80) {
    int n = tid / 5, c = tid % 5;
    float acc = sLtib[c];
    for (int j = 0; j < 64; ++j) acc += sLtiw[c*64 + j] * sZ0[n*64 + j];
    sLog[n*8 + c] = acc;
  }
  __syncthreads();
  if (tid < NB) {
    float m = -1e30f;
    #pragma unroll
    for (int c = 0; c < 5; ++c) m = fmaxf(m, sLog[tid*8 + c]);
    float e[5], sum = 0.f;
    #pragma unroll
    for (int c = 0; c < 5; ++c) { e[c] = __expf(sLog[tid*8 + c] - m); sum += e[c]; }
    float inv = fast_rcp(sum);
    #pragma unroll
    for (int c = 0; c < 5; ++c) st0[(size_t)(b0 + tid) * 136 + c] = e[c] * inv;
  }
}

// ======================= Tsit5 ODE integrator =======================
#define ODE_SMEM 324800u

__global__ __launch_bounds__(NTHR)
void ode_kernel(const float* __restrict__ ts, const float* __restrict__ scaleg,
                const float* __restrict__ w0, const float* __restrict__ b0g,
                const float* __restrict__ w1, const float* __restrict__ b1g,
                const float* __restrict__ w2, const float* __restrict__ b2g,
                const float* __restrict__ htbwg, const float* __restrict__ htbbg,
                const float* __restrict__ st0,
                float* __restrict__ out_y, float* __restrict__ out_h)
{
  extern __shared__ char smem[];
  size_t _off = 0;
  CARVE(_Float16, sW0,  WW*HH);   // 64 KB  (mlp_w0 cols 1..128; t folded)
  CARVE(_Float16, sW1,  WW*WW);   // 128 KB
  CARVE(_Float16, sW2,  HH*WW);   // 64 KB
  CARVE(float,    sW0c0, WW);     // mlp_w0 column 0
  CARVE(float,    sB0,  WW);
  CARVE(float,    sB1,  WW);
  CARVE(float,    sB2,  HH);
  CARVE(float,    sHtbw, HH);
  CARVE(_Float16, sXT0, NB*HH);   // stage input h, column-major [n][128]
  CARVE(_Float16, sXT1, NB*WW);   // layer-0 activations [n][256]
  CARVE(_Float16, sXT2, NB*WW);   // layer-1 activations [n][256]
  CARVE(float,    sYh,  NB*HH);   // base state h (f32)
  CARVE(float,    sYs,  NB*8);    // base SEIR (5 used)
  CARVE(float,    sYsc, NB*8);    // stage SEIR argument
  CARVE(_Float16, sKh,  6*NB*HH); // k1..k6, h-part (f16)
  CARVE(float,    sKs,  6*NB*8);  // k1..k6, SEIR part (f32)
  CARVE(float,    sRed, NB*16);   // beta dot partials
  CARVE(float,    sTst, NB);      // per-batch stage time
  CARVE(float,    sHs,  NB);      // per-batch substep h
  CARVE(float,    sT0,  NB);      // per-batch carry time

  const int tid  = threadIdx.x, lane = tid & 31;
  const int wave = __builtin_amdgcn_readfirstlane(tid >> 5);  // scalar wave id
  const int b0 = blockIdx.x * NB;
  const float scaleV = scaleg[0] * 0.1f;   // dh = scale * 0.1 * tanh(...)
  const float htbbV  = htbbg[0];

  // ---- stage weights + state into LDS ----
  for (int i = tid; i < WW; i += NTHR) { sW0c0[i] = w0[i*129]; sB0[i] = b0g[i]; sB1[i] = b1g[i]; }
  for (int i = tid; i < WW*HH; i += NTHR) {
    int m = i >> 7, k = i & 127;
    sW0[i] = (_Float16)w0[m*129 + 1 + k];
  }
  for (int i = tid; i < WW*WW; i += NTHR) sW1[i] = (_Float16)w1[i];
  for (int i = tid; i < HH*WW; i += NTHR) sW2[i] = (_Float16)w2[i];
  for (int i = tid; i < HH; i += NTHR) { sB2[i] = b2g[i]; sHtbw[i] = htbwg[i]; }
  for (int i = tid; i < NB*HH; i += NTHR) {
    int n = i >> 7, j = i & 127;
    sYh[i] = st0[(size_t)(b0 + n) * 136 + 5 + j];
  }
  if (tid < NB*8) {
    int n = tid >> 3, c = tid & 7;
    sYs[tid] = (c < 5) ? st0[(size_t)(b0 + n) * 136 + c] : 0.f;
  }
  if (tid < NB) sT0[tid] = 0.f;
  __syncthreads();

  for (int jT = 0; jT < TT; ++jT) {
    if (tid < NB) {
      float tn = ts[(size_t)(b0 + tid) * TT + jT];
      sHs[tid] = (tn - sT0[tid]) * 0.5f;    // K_SUB = 2
    }
    __syncthreads();

    for (int sub = 0; sub < 2; ++sub) {
      for (int st = 0; st < 6; ++st) {
        // ---- build stage argument  y + h * sum(A[st][q] * k_q) ----
        if (tid < NB) sTst[tid] = sT0[tid] + ((float)sub + c_C[st]) * sHs[tid];
        for (int i = tid; i < NB*HH; i += NTHR) {
          int n = i >> 7;
          float h = sHs[n], acc = sYh[i];
          for (int q = 0; q < st; ++q)
            acc += h * c_A[st][q] * (float)sKh[q*NB*HH + i];
          sXT0[i] = (_Float16)acc;
        }
        if (tid < NB*8) {
          int n = tid >> 3;
          float h = sHs[n], acc = sYs[tid];
          for (int q = 0; q < st; ++q)
            acc += h * c_A[st][q] * sKs[q*NB*8 + tid];
          sYsc[tid] = acc;
        }
        __syncthreads();

        // ---- beta dot partials + MLP layer 0 (GEMM 256x16, K=128) ----
        {
          int n = tid >> 4, seg = tid & 15;
          const _Float16* xr = sXT0 + n*HH + seg*8;
          float p = 0.f;
          #pragma unroll
          for (int j = 0; j < 8; ++j) p += sHtbw[seg*8 + j] * (float)xr[j];
          sRed[tid] = p;
        }
        for (int mt = wave; mt < 16; mt += NWAVE) {
          v8f acc = wmma_tile(sW0, HH, mt * 16, sXT0, 4);
          int n = lane & 15, row0 = mt * 16 + ((lane >> 4) << 3);
          float tn = sTst[n];
          v8h hv;
          #pragma unroll
          for (int v = 0; v < 8; ++v) {
            int row = row0 + v;
            hv[v] = (_Float16)softplus(acc[v] + sB0[row] + tn * sW0c0[row]);
          }
          *(v8h*)(sXT1 + n*WW + row0) = hv;
        }
        __syncthreads();

        // ---- SEIR derivatives (needs beta) + MLP layer 1 (K=256) ----
        if (tid < NB) {
          float d = htbbV;
          #pragma unroll
          for (int g = 0; g < 16; ++g) d += sRed[tid*16 + g];
          float beta = sigm(0.01f * d);
          float S = sYsc[tid*8 + 0], E = sYsc[tid*8 + 1];
          float I = sYsc[tid*8 + 2], A = sYsc[tid*8 + 3];
          float LL  = 0.5f * I + A;           // ee=0, q=0.5, dd=1
          float bSL = beta * S * LL;
          float kE  = 0.526f * E;
          float* k = sKs + st*NB*8 + tid*8;
          k[0] = -bSL;
          k[1] = bSL - kE;
          k[2] = 0.667f * kE - 0.244f * I;
          k[3] = (1.0f - 0.667f) * kE - 0.244f * A;
          k[4] = 0.98f * 0.244f * I + 0.244f * A;
        }
        for (int mt = wave; mt < 16; mt += NWAVE) {
          v8f acc = wmma_tile(sW1, WW, mt * 16, sXT1, 8);
          int n = lane & 15, row0 = mt * 16 + ((lane >> 4) << 3);
          v8h hv;
          #pragma unroll
          for (int v = 0; v < 8; ++v)
            hv[v] = (_Float16)softplus(acc[v] + sB1[row0 + v]);
          *(v8h*)(sXT2 + n*WW + row0) = hv;
        }
        __syncthreads();

        // ---- MLP layer 2 (128x16, K=256) -> k_h[st] ----
        for (int mt = wave; mt < 8; mt += NWAVE) {
          v8f acc = wmma_tile(sW2, WW, mt * 16, sXT2, 8);
          int n = lane & 15, row0 = mt * 16 + ((lane >> 4) << 3);
          v8h hv;
          #pragma unroll
          for (int v = 0; v < 8; ++v)
            hv[v] = (_Float16)(scaleV * fast_tanh(1e-4f * (acc[v] + sB2[row0 + v])));
          *(v8h*)(sKh + st*NB*HH + n*HH + row0) = hv;
        }
        __syncthreads();
      } // stages

      // ---- y += h * sum(B_i * k_i) ----
      for (int i = tid; i < NB*HH; i += NTHR) {
        int n = i >> 7;
        float h = sHs[n], acc = sYh[i];
        #pragma unroll
        for (int q = 0; q < 6; ++q) acc += h * c_B[q] * (float)sKh[q*NB*HH + i];
        sYh[i] = acc;
      }
      if (tid < NB*8) {
        int n = tid >> 3;
        float h = sHs[n], acc = sYs[tid];
        #pragma unroll
        for (int q = 0; q < 6; ++q) acc += h * c_B[q] * sKs[q*NB*8 + tid];
        sYs[tid] = acc;
      }
      __syncthreads();
    } // sub

    // ---- save trajectory, advance carry time ----
    for (int i = tid; i < NB*HH; i += NTHR) {
      int n = i >> 7, j = i & 127;
      out_h[((size_t)(b0 + n) * TT + jT) * HH + j] = sYh[i];
    }
    if (tid < NB*8) {
      int n = tid >> 3, c = tid & 7;
      if (c < 5) out_y[((size_t)(b0 + n) * TT + jT) * 5 + c] = sYs[tid];
    }
    if (tid < NB) sT0[tid] = ts[(size_t)(b0 + tid) * TT + jT];
    __syncthreads();
  }
}

// ======================= host launch =======================
extern "C" void kernel_launch(void* const* d_in, const int* in_sizes, int n_in,
                              void* d_out, int out_size, void* d_ws, size_t ws_size,
                              hipStream_t stream) {
  const float* ts    = (const float*)d_in[0];
  const float* ys    = (const float*)d_in[1];
  const float* scale = (const float*)d_in[2];
  const float* w0    = (const float*)d_in[3];
  const float* b0    = (const float*)d_in[4];
  const float* w1    = (const float*)d_in[5];
  const float* b1    = (const float*)d_in[6];
  const float* w2    = (const float*)d_in[7];
  const float* b2    = (const float*)d_in[8];
  const float* gwih  = (const float*)d_in[9];
  const float* gwhh  = (const float*)d_in[10];
  const float* gb    = (const float*)d_in[11];
  const float* gbn   = (const float*)d_in[12];
  const float* htlw  = (const float*)d_in[13];
  const float* htlb  = (const float*)d_in[14];
  const float* htbw  = (const float*)d_in[15];
  const float* htbb  = (const float*)d_in[16];
  const float* ltiw  = (const float*)d_in[17];
  const float* ltib  = (const float*)d_in[18];
  const float* lthw  = (const float*)d_in[19];
  const float* lthb  = (const float*)d_in[20];

  float* out_y = (float*)d_out;              // (64,64,5)
  float* out_h = out_y + 64 * 64 * 5;        // (64,64,128)
  float* out_z = out_h + 64 * 64 * 128;      // (64,64)
  float* st0   = (float*)d_ws;               // (64,136) f32 state0 scratch

  enc_kernel<<<dim3(4), dim3(NTHR), ENC_SMEM, stream>>>(
      ts, ys, gwih, gwhh, gb, gbn, htlw, htlb, ltiw, ltib, lthw, lthb, out_z, st0);
  ode_kernel<<<dim3(4), dim3(NTHR), ODE_SMEM, stream>>>(
      ts, scale, w0, b0, w1, b1, w2, b2, htbw, htbb, st0, out_y, out_h);
}
